// LOHCGNN_for_MP_BP_32315333935677
// MI455X (gfx1250) — matrix operand order repaired and checked by reading, not compiled, not run on hardware
//
#include <hip/hip_runtime.h>
#include <hip/hip_bf16.h>

// MI455X / gfx1250, wave32. GEMMs via v_wmma_f32_16x16x32_f16 (f16 in, f32 acc).
// Edge-gated conv fused kernel: workgroup stages this layer's 7 packed weight
// matrices (56KB f16) into LDS once (CDNA5: 320KB LDS/WG), each wave owns a
// 16-edge tile (x_i/x_j/e rows gathered to LDS as f16), runs 56 WMMAs/tile with
// B-fragments batched from LDS, then gate+message+scatter-add in f32.

typedef __attribute__((ext_vector_type(16))) _Float16 v16h;
typedef __attribute__((ext_vector_type(8)))  _Float16 v8h;
typedef __attribute__((ext_vector_type(8)))  float    v8f;

#define LDSS 72               // halves per staged tile row (64 + pad to spread LDS banks)
#define WPB  8                // waves per block (256 threads)

__device__ __forceinline__ int imin(int a, int b) { return a < b ? a : b; }

__device__ __forceinline__ v8f wmma_f16(v16h a, v16h b, v8f c) {
  // (neg_a, A, neg_b, B, c_mod, C, reuse_a, reuse_b)
  return __builtin_amdgcn_wmma_f32_16x16x32_f16(false, a, false, b, (short)0, c, false, false);
}

// B fragment from packed layout: p[((nt*KS+ks)*32 + lane)*16 + j] (global or LDS ptr)
__device__ __forceinline__ v16h load_bfrag(const _Float16* p, int nt, int ks, int KS, int lane) {
  return *(const v16h*)(p + (size_t)(((nt * KS) + ks) * 32 + lane) * 16);
}

// A fragment (16x32 f16, ISA layout) from an LDS-staged 16 x (KS*32) tile, row stride LDSS.
__device__ __forceinline__ v16h load_afrag(const _Float16* tile, int m, int half, int ks) {
  const _Float16* p = tile + m * LDSS + ks * 32 + 8 * half;
  v8h lo = *(const v8h*)(p);
  v8h hi = *(const v8h*)(p + 16);
  v16h a;
#pragma unroll
  for (int j = 0; j < 8; ++j) { a[j] = lo[j]; a[8 + j] = hi[j]; }
  return a;
}

// Stage 16 rows x 64 f32 (gathered via gidx, or contiguous when gidx==nullptr)
// into an f16 LDS tile. Coalesced float4 global loads, b64 LDS stores.
__device__ __forceinline__ void stage64(_Float16* tile, const float* __restrict__ base,
                                        const int* __restrict__ gidx, int row0, int maxRow,
                                        int lane) {
#pragma unroll
  for (int t = 0; t < 8; ++t) {
    int r  = t * 2 + (lane >> 4);
    int c  = (lane & 15) * 4;
    int rr = imin(row0 + r, maxRow - 1);
    int grow = gidx ? gidx[rr] : rr;
    float4 v = *(const float4*)(base + (size_t)grow * 64 + c);
    _Float16* d = tile + r * LDSS + c;
    d[0] = (_Float16)v.x; d[1] = (_Float16)v.y; d[2] = (_Float16)v.z; d[3] = (_Float16)v.w;
  }
}

// ---- weight packing: W[Kin][64] f32 -> f16 B-fragment layout (zero-padded K) ----
__global__ void pack_w_kernel(const float* __restrict__ W, _Float16* __restrict__ dst,
                              int Kin, int KS) {
  int t = blockIdx.x * 256 + threadIdx.x;
  int total = 4 * KS * 32 * 16;
  if (t >= total) return;
  int j    = t & 15;
  int lane = (t >> 4) & 31;
  int rest = t >> 9;
  int ks   = rest % KS;
  int nt   = rest / KS;
  int half = lane >> 4, n = lane & 15;
  int kk   = ks * 32 + (j < 8 ? 8 * half + j : 16 + 8 * half + (j - 8));  // B layout: j<8 -> K 8h+j, j>=8 -> K 16+8h+(j-8)
  float v  = (kk < Kin) ? W[kk * 64 + nt * 16 + n] : 0.0f;
  dst[t]   = (_Float16)v;
}

// ---- embedding GEMM: out[rows,64] = A[rows,Kin] @ W + bias (Kin = 16 or 64) ----
__global__ void embed_gemm_kernel(const float* __restrict__ A, int rows, int Kin,
                                  const _Float16* __restrict__ pW,
                                  const float* __restrict__ bias,
                                  float* __restrict__ out) {
  __shared__ _Float16 wlds[4096];
  __shared__ _Float16 smem[WPB * 16 * LDSS];
  int lane = threadIdx.x & 31;
  int wave = threadIdx.x >> 5;
  int row0 = (blockIdx.x * WPB + wave) * 16;
  int KS = (Kin == 64) ? 2 : 1;
  // cooperative weight stage (2048*KS halves = 256*KS uint4)
  {
    const uint4* s = (const uint4*)pW;
    uint4* d4 = (uint4*)wlds;
    for (int t = threadIdx.x; t < 256 * KS; t += 256) d4[t] = s[t];
  }
  bool active = row0 < rows;
  _Float16* T = smem + wave * 16 * LDSS;
  if (active) {
    if (Kin == 64) {
      stage64(T, A, nullptr, row0, rows, lane);
    } else {                                    // Kin == 16, zero-pad K to 32
#pragma unroll
      for (int t = 0; t < 12; ++t) {            // zero halves [16..63] (16 rows x 24 dwords)
        int i = t * 32 + lane;
        int r = i / 24, cd = i % 24;
        *(unsigned int*)(T + r * LDSS + 16 + cd * 2) = 0u;
      }
#pragma unroll
      for (int t = 0; t < 2; ++t) {
        int fl = t * 128 + lane * 4;
        int r = fl >> 4, c = fl & 15;
        int rr = imin(row0 + r, rows - 1);
        float4 v = *(const float4*)(A + (size_t)rr * 16 + c);
        _Float16* d = T + r * LDSS + c;
        d[0] = (_Float16)v.x; d[1] = (_Float16)v.y; d[2] = (_Float16)v.z; d[3] = (_Float16)v.w;
      }
    }
  }
  __syncthreads();
  if (!active) return;

  int ml = lane & 15, half = lane >> 4;
  v8f acc[4] = {};
  for (int ks = 0; ks < KS; ++ks) {
    v16h a = load_afrag(T, ml, half, ks);
    v16h B0 = load_bfrag(wlds, 0, ks, KS, lane);
    v16h B1 = load_bfrag(wlds, 1, ks, KS, lane);
    v16h B2 = load_bfrag(wlds, 2, ks, KS, lane);
    v16h B3 = load_bfrag(wlds, 3, ks, KS, lane);
    acc[0] = wmma_f16(a, B0, acc[0]);
    acc[1] = wmma_f16(a, B1, acc[1]);
    acc[2] = wmma_f16(a, B2, acc[2]);
    acc[3] = wmma_f16(a, B3, acc[3]);
  }
#pragma unroll
  for (int nt = 0; nt < 4; ++nt) {
    int col = nt * 16 + ml;
    float bv = bias[col];
#pragma unroll
    for (int v = 0; v < 8; ++v) {
      int r = row0 + v + 8 * half;              // C layout: M = v + 8*half, N = lane&15
      if (r < rows) out[(size_t)r * 64 + col] = acc[nt][v] + bv;
    }
  }
}

// ---- fused edge-gated conv over a 16-edge tile per wave; weights in LDS ----
__global__ void edge_gated_kernel(const float* __restrict__ x,   // [Nn,64] node feats (ro)
                                  float* __restrict__ e,         // [Ne,64] edge feats (rw)
                                  const int* __restrict__ src, const int* __restrict__ dst,
                                  int nE,
                                  float* __restrict__ xacc,      // [Nn,64] pre-zeroed accum
                                  const _Float16* __restrict__ pW, // 7 contiguous packed mats
                                  const float* __restrict__ bg, const float* __restrict__ bn,
                                  const float* __restrict__ be) {
  __shared__ _Float16 wlds[7 * 4096];            // 56KB packed weights (Wg_x..We_e)
  __shared__ _Float16 tiles[WPB * 3 * 16 * LDSS];
  int lane  = threadIdx.x & 31;
  int wave  = threadIdx.x >> 5;
  int edge0 = (blockIdx.x * WPB + wave) * 16;
  // cooperative weight stage: 7*4096 halves = 3584 uint4, 14 per thread
  {
    const uint4* s = (const uint4*)pW;
    uint4* d4 = (uint4*)wlds;
#pragma unroll
    for (int t = 0; t < 14; ++t) d4[t * 256 + threadIdx.x] = s[t * 256 + threadIdx.x];
  }
  bool active = edge0 < nE;
  _Float16* XI = tiles + wave * (3 * 16 * LDSS);
  _Float16* XJ = XI + 16 * LDSS;
  _Float16* EE = XJ + 16 * LDSS;
  if (active) {
    stage64(XI, x, dst, edge0, nE, lane);        // x_i = x[dst]
    stage64(XJ, x, src, edge0, nE, lane);        // x_j = x[src]
    stage64(EE, e, nullptr, edge0, nE, lane);    // e rows (contiguous)
  }
  __syncthreads();
  if (!active) return;

  int ml = lane & 15, half = lane >> 4;
  v16h aXI[2], aXJ[2], aEE[2];
#pragma unroll
  for (int ks = 0; ks < 2; ++ks) {
    aXI[ks] = load_afrag(XI, ml, half, ks);
    aXJ[ks] = load_afrag(XJ, ml, half, ks);
    aEE[ks] = load_afrag(EE, ml, half, ks);
  }

#pragma unroll
  for (int nt = 0; nt < 4; ++nt) {
    v8f G = {}, Mv = {}, R = {};
#pragma unroll
    for (int ks = 0; ks < 2; ++ks) {
      // batch the 7 B-fragment LDS loads so one wait covers the 7-WMMA chain
      v16h B0 = load_bfrag(wlds + 0 * 4096, nt, ks, 2, lane);  // Wg_x
      v16h B1 = load_bfrag(wlds + 1 * 4096, nt, ks, 2, lane);  // Wg_e
      v16h B2 = load_bfrag(wlds + 2 * 4096, nt, ks, 2, lane);  // Wn_x
      v16h B3 = load_bfrag(wlds + 3 * 4096, nt, ks, 2, lane);  // Wn_e
      v16h B4 = load_bfrag(wlds + 4 * 4096, nt, ks, 2, lane);  // We_r
      v16h B5 = load_bfrag(wlds + 5 * 4096, nt, ks, 2, lane);  // We_c
      v16h B6 = load_bfrag(wlds + 6 * 4096, nt, ks, 2, lane);  // We_e
      G  = wmma_f16(aXI[ks], B0, G);
      G  = wmma_f16(aEE[ks], B1, G);
      Mv = wmma_f16(aXJ[ks], B2, Mv);
      Mv = wmma_f16(aEE[ks], B3, Mv);
      R  = wmma_f16(aXJ[ks], B4, R);
      R  = wmma_f16(aXI[ks], B5, R);
      R  = wmma_f16(aEE[ks], B6, R);
    }
    int col = nt * 16 + ml;
    float bgv = bg[col], bnv = bn[col], bev = be[col];
#pragma unroll
    for (int v = 0; v < 8; ++v) {
      int er = edge0 + v + 8 * half;
      if (er < nE) {
        float gate = 1.0f / (1.0f + __expf(-(G[v] + bgv)));
        float msg  = gate * (Mv[v] + bnv);
        atomicAdd(xacc + (size_t)dst[er] * 64 + col, msg);   // segment_sum over dst
        size_t eo = (size_t)er * 64 + col;
        e[eo] += R[v] + bev;                                 // e = e + e_up (residual)
      }
    }
  }
}

__global__ void add_inplace_kernel(float* __restrict__ x, const float* __restrict__ a, size_t n) {
  size_t i = (size_t)blockIdx.x * 256 + threadIdx.x;
  if (i < n) x[i] += a[i];
}

__global__ void pool_kernel(const float* __restrict__ h, const int* __restrict__ batch,
                            float* __restrict__ pooled, float* __restrict__ cnt, int N) {
  int t = blockIdx.x * 256 + threadIdx.x;
  if (t >= N * 64) return;
  int a = t >> 6, c = t & 63;
  int b = batch[a];
  atomicAdd(pooled + b * 64 + c, h[t]);
  if (c == 0) atomicAdd(cnt + b, 1.0f);
}

__global__ void mlp_kernel(const float* __restrict__ pooled, const float* __restrict__ cnt,
                           const float* __restrict__ W1, const float* __restrict__ b1,
                           const float* __restrict__ W2, const float* __restrict__ b2,
                           float* __restrict__ out) {
  int g = threadIdx.x;
  if (g >= 128) return;
  float inv = 1.0f / fmaxf(cnt[g], 1.0f);
  float o = b2[0];
  for (int j = 0; j < 64; ++j) {
    float s = b1[j];
    for (int c = 0; c < 64; ++c) s += pooled[g * 64 + c] * inv * W1[c * 64 + j];
    o += fmaxf(s, 0.0f) * W2[j];
  }
  out[g] = o;
}

extern "C" void kernel_launch(void* const* d_in, const int* in_sizes, int n_in,
                              void* d_out, int out_size, void* d_ws, size_t ws_size,
                              hipStream_t stream) {
  // Inputs in setup_inputs() recursive dict (insertion) order.
  const float* atom_x  = (const float*)d_in[0];
  const int*   atom_ei = (const int*)  d_in[1];
  const float* atom_ea = (const float*)d_in[2];
  const int*   atom_bt = (const int*)  d_in[3];
  const float* line_x  = (const float*)d_in[4];
  const int*   line_ei = (const int*)  d_in[5];
  const float* line_ea = (const float*)d_in[6];
  const float* node_W  = (const float*)d_in[7];
  const float* node_b  = (const float*)d_in[8];
  const float* edge_W  = (const float*)d_in[9];
  const float* edge_b  = (const float*)d_in[10];
  const float* ledge_W = (const float*)d_in[11];
  const float* ledge_b = (const float*)d_in[12];
  const float* aW[7]; const float* lW[7];
  for (int w = 0; w < 7; ++w) aW[w] = (const float*)d_in[13 + w];   // Wg_x,Wg_e,Wn_x,Wn_e,We_r,We_c,We_e
  const float* a_bg = (const float*)d_in[20];
  const float* a_bn = (const float*)d_in[21];
  const float* a_be = (const float*)d_in[22];
  for (int w = 0; w < 7; ++w) lW[w] = (const float*)d_in[23 + w];
  const float* l_bg = (const float*)d_in[30];
  const float* l_bn = (const float*)d_in[31];
  const float* l_be = (const float*)d_in[32];
  const float* mW1 = (const float*)d_in[33];
  const float* mb1 = (const float*)d_in[34];
  const float* mW2 = (const float*)d_in[35];
  const float* mb2 = (const float*)d_in[36];

  const int N  = in_sizes[3];       // atoms (50000)
  const int E  = in_sizes[1] / 2;   // atom edges (200000)
  const int LN = in_sizes[4] / 16;  // line nodes (200000)
  const int LE = in_sizes[5] / 2;   // line edges (400000)

  // workspace layout (~269 MB): all regions written before read each call
  float* ws     = (float*)d_ws;
  float* h      = ws;
  float* e      = h  + (size_t)N  * 64;
  float* l      = e  + (size_t)E  * 64;
  float* le     = l  + (size_t)LN * 64;
  float* xacc   = le + (size_t)LE * 64;          // sized for max(LN, N)
  float* pooled = xacc + (size_t)LN * 64;
  float* cnt    = pooled + 128 * 64;
  _Float16* pk  = (_Float16*)(cnt + 128);        // 256B-aligned (all regions mult. of 64 floats)
  auto PS = [&](int s) { return pk + (size_t)s * 4096; };

  // pack weights into f16 B-fragment layout.
  // Slots: 0 node_emb, 1 edge_emb, 2 line_edge_emb,
  //        3  + i*7 + w : atom conv, layer i, matrix w (7 contiguous per layer)
  //        17 + i*7 + w : line conv, layer i, matrix w
  pack_w_kernel<<<16, 256, 0, stream>>>(node_W,  PS(0), 64, 2);
  pack_w_kernel<<< 8, 256, 0, stream>>>(edge_W,  PS(1), 16, 1);
  pack_w_kernel<<< 8, 256, 0, stream>>>(ledge_W, PS(2), 16, 1);
  for (int w = 0; w < 7; ++w)
    for (int i = 0; i < 2; ++i) {
      pack_w_kernel<<<16, 256, 0, stream>>>(aW[w] + (size_t)i * 4096, PS(3  + i * 7 + w), 64, 2);
      pack_w_kernel<<<16, 256, 0, stream>>>(lW[w] + (size_t)i * 4096, PS(17 + i * 7 + w), 64, 2);
    }

  auto grid16 = [](int rows) { return ((rows + 15) / 16 + WPB - 1) / WPB; };

  // embeddings
  embed_gemm_kernel<<<grid16(N),  256, 0, stream>>>(atom_x,  N,  64, PS(0), node_b,  h);
  embed_gemm_kernel<<<grid16(E),  256, 0, stream>>>(atom_ea, E,  16, PS(1), edge_b,  e);
  embed_gemm_kernel<<<grid16(LN), 256, 0, stream>>>(line_x,  LN, 16, PS(1), edge_b,  l);  // edge_emb for line nodes
  embed_gemm_kernel<<<grid16(LE), 256, 0, stream>>>(line_ea, LE, 16, PS(2), ledge_b, le);

  for (int i = 0; i < 2; ++i) {
    // line graph (independent of atom graph; layer order preserved per graph)
    hipMemsetAsync(xacc, 0, (size_t)LN * 64 * sizeof(float), stream);
    edge_gated_kernel<<<grid16(LE), 256, 0, stream>>>(
        l, le, line_ei, line_ei + LE, LE, xacc, PS(17 + i * 7),
        l_bg + i * 64, l_bn + i * 64, l_be + i * 64);
    add_inplace_kernel<<<(int)(((size_t)LN * 64 + 255) / 256), 256, 0, stream>>>(l, xacc, (size_t)LN * 64);
    // atom graph
    hipMemsetAsync(xacc, 0, (size_t)N * 64 * sizeof(float), stream);
    edge_gated_kernel<<<grid16(E), 256, 0, stream>>>(
        h, e, atom_ei, atom_ei + E, E, xacc, PS(3 + i * 7),
        a_bg + i * 64, a_bn + i * 64, a_be + i * 64);
    add_inplace_kernel<<<(int)(((size_t)N * 64 + 255) / 256), 256, 0, stream>>>(h, xacc, (size_t)N * 64);
  }

  // mean pool + readout MLP
  hipMemsetAsync(pooled, 0, (128 * 64 + 128) * sizeof(float), stream);
  pool_kernel<<<(N * 64 + 255) / 256, 256, 0, stream>>>(h, atom_bt, pooled, cnt, N);
  mlp_kernel<<<1, 128, 0, stream>>>(pooled, cnt, mW1, mb1, mW2, mb2, (float*)d_out);

  (void)n_in; (void)out_size; (void)ws_size;
}
